// minLSTMPar_47605417509426
// MI455X (gfx1250) — matrix operand order repaired and checked by reading, not compiled, not run on hardware
//
#include <hip/hip_runtime.h>
#include <hip/hip_bf16.h>
#include <math.h>

// Problem constants
#define Bb 16
#define Tt 4096
#define Dd 512
#define Hh 512
#define Mrows (Bb*Tt)          // 65536
#define CHUNKS 64
#define CLEN  (Tt/CHUNKS)      // 64

typedef __attribute__((ext_vector_type(16))) __bf16 v16bf;
typedef __attribute__((ext_vector_type(8)))  __bf16 v8bf;
typedef __attribute__((ext_vector_type(8)))  float  v8f;

__device__ __forceinline__ float softplusf(float z) {
    // numerically stable softplus
    return fmaxf(z, 0.f) + log1pf(expf(-fabsf(z)));
}
__device__ __forceinline__ float log_gf(float x) {
    return (x >= 0.f) ? logf(x + 0.5f) : -softplusf(-x);
}

// ---------------------------------------------------------------------------
// Kernel 0: bf16 hi/lo split of the three weight matrices, no transpose.
// W[p][h][d] (f32, row-major) -> Wt_hi/Wt_lo[p][h][d] (bf16) so that WMMA
// B-fragments (lane = N(h), elements = K(d) in runs of 8) load contiguously.
// ---------------------------------------------------------------------------
__global__ void prep_w(const float* __restrict__ Wf, const float* __restrict__ Wi,
                       const float* __restrict__ Wh,
                       __bf16* __restrict__ whi, __bf16* __restrict__ wlo) {
    int idx = blockIdx.x * blockDim.x + threadIdx.x;
    if (idx >= 3 * Dd * Hh) return;
    int p = idx / (Dd * Hh);
    int r = idx % (Dd * Hh);
    int h = r / Dd;
    int d = r % Dd;
    const float* W = (p == 0) ? Wf : (p == 1) ? Wi : Wh;
    float v = W[(size_t)h * Dd + d];
    __bf16 hi = (__bf16)v;
    __bf16 lo = (__bf16)(v - (float)hi);
    whi[idx] = hi;
    wlo[idx] = lo;
}

// ---------------------------------------------------------------------------
// Kernel 1: fused 3-way GEMM (bf16x3 split WMMA) + gate activations +
// hidden-axis prefix sum.
//   writes v    = log_i + log g(lin_h)        -> d_out (overwritten later)
//   writes scum = cumsum_h(log_f) (inclusive) -> ws
// Grid: Mrows/16 blocks of 256 threads (8 waves). Each wave owns 4 N-tiles,
// computing f/i/h accumulators simultaneously so activations fuse in-register.
// ---------------------------------------------------------------------------
#define XPITCH 520   // 16B-aligned row pitch, staggers LDS banks
#define LFPITCH 513

__global__ void gemm_act(const float* __restrict__ x,
                         const __bf16* __restrict__ whi,
                         const __bf16* __restrict__ wlo,
                         const float* __restrict__ bf_, const float* __restrict__ bi_,
                         const float* __restrict__ bh_,
                         float* __restrict__ vout,    // [Mrows][Hh]  (= d_out)
                         float* __restrict__ scum)    // [Mrows][Hh]  (workspace)
{
    extern __shared__ char smem[];
    __bf16* xh  = (__bf16*)smem;                 // 16*520*2 = 16640 B
    __bf16* xl  = (__bf16*)(smem + 16640);       // 16640 B
    float*  lf  = (float*)(smem + 33280);        // 16*513*4 = 32832 B
    float*  tot = (float*)(smem + 66112);        // 16*16*4  = 1024 B

    const int tid = threadIdx.x;
    const int m0  = blockIdx.x * 16;

    // Stage x tile (16 rows x 512) as bf16 hi/lo in LDS (coalesced f32 loads).
    for (int i = tid; i < 16 * Dd; i += 256) {
        int r = i >> 9, c = i & (Dd - 1);
        float v = x[(size_t)(m0 + r) * Dd + c];
        __bf16 hi = (__bf16)v;
        xh[r * XPITCH + c] = hi;
        xl[r * XPITCH + c] = (__bf16)(v - (float)hi);
    }
    __syncthreads();

    const int wave = tid >> 5;
    const int lane = tid & 31;
    const int arow = lane & 15;       // A: lanes 0-15 -> M=lane, 16-31 -> M=lane-16
    const int ksel = lane >> 4;       // A K-chunk select per ISA layout

    for (int nti = 0; nti < 4; ++nti) {
        const int nt = wave + nti * 8;
        const int nb = nt * 16;
        v8f acc[3] = {};
        for (int kt = 0; kt < Dd / 32; ++kt) {
            // A fragments (16x32 bf16) from LDS per documented lane layout:
            // lane<16: K = {0..7, 16..23}; lane>=16: K = {8..15, 24..31}
            const int kbase = kt * 32 + ksel * 8;
            const __bf16* ph = &xh[arow * XPITCH + kbase];
            const __bf16* pl = &xl[arow * XPITCH + kbase];
            v16bf Ah, Al;
            ((v8bf*)&Ah)[0] = *(const v8bf*)(ph);
            ((v8bf*)&Ah)[1] = *(const v8bf*)(ph + 16);
            ((v8bf*)&Al)[0] = *(const v8bf*)(pl);
            ((v8bf*)&Al)[1] = *(const v8bf*)(pl + 16);
            // B fragments (32x16 bf16): lane = N(h), elements = K(d) in same order as A
            const int d = kt * 32 + ksel * 8;
            const size_t bidx = (size_t)(nb + arow) * Dd + d;
#pragma unroll
            for (int p = 0; p < 3; ++p) {
                const __bf16* pbh = whi + (size_t)p * Dd * Hh + bidx;
                const __bf16* pbl = wlo + (size_t)p * Dd * Hh + bidx;
                v16bf Bh, Bl;
                ((v8bf*)&Bh)[0] = *(const v8bf*)(pbh);
                ((v8bf*)&Bh)[1] = *(const v8bf*)(pbh + 16);
                ((v8bf*)&Bl)[0] = *(const v8bf*)(pbl);
                ((v8bf*)&Bl)[1] = *(const v8bf*)(pbl + 16);
                // bf16x3 split product: hi*hi + hi*lo + lo*hi, f32 accumulate
                acc[p] = __builtin_amdgcn_wmma_f32_16x16x32_bf16(
                            false, Ah, false, Bh, (short)0, acc[p], false, false);
                acc[p] = __builtin_amdgcn_wmma_f32_16x16x32_bf16(
                            false, Ah, false, Bl, (short)0, acc[p], false, false);
                acc[p] = __builtin_amdgcn_wmma_f32_16x16x32_bf16(
                            false, Al, false, Bh, (short)0, acc[p], false, false);
            }
        }
        // Epilogue: C layout -> element r of lane l is (M = r + (l>=16 ? 8:0),
        // N = l&15). f/i/h accumulators align elementwise -> fuse activations.
        const int n    = nb + (lane & 15);
        const int mofs = (lane >> 4) * 8;
        const float bfv = bf_[n], biv = bi_[n], bhv = bh_[n];
#pragma unroll
        for (int r = 0; r < 8; ++r) {
            const int m = mofs + r;
            float lfv = acc[0][r] + bfv;
            float liv = acc[1][r] + biv;
            float lhv = acc[2][r] + bhv;
            float diff  = softplusf(-lfv) - softplusf(-liv);
            float logfv = -softplusf(diff);                  // log_f
            float vval  = -softplusf(-diff) + log_gf(lhv);   // log_i + log g
            vout[(size_t)(m0 + m) * Hh + n] = vval;
            lf[m * LFPITCH + n] = logfv;
        }
    }
    __syncthreads();

    // Inclusive prefix sum of log_f over h (512 per row), 16 segs x 32/row.
    {
        const int r = tid >> 4, seg = tid & 15, base = seg * 32;
        float s = 0.f;
        for (int j = 0; j < 32; ++j) s += lf[r * LFPITCH + base + j];
        tot[r * 16 + seg] = s;
    }
    __syncthreads();
    {
        const int r = tid >> 4, seg = tid & 15, base = seg * 32;
        float run = 0.f;
        for (int k = 0; k < seg; ++k) run += tot[r * 16 + k];
        float* dst = scum + (size_t)(m0 + r) * Hh + base;
        for (int j = 0; j < 32; ++j) {
            run += lf[r * LFPITCH + base + j];
            dst[j] = run;
        }
    }
}

// ---------------------------------------------------------------------------
// Chunked cumulative-logsumexp over time (the only sequential dependency).
// Chain element e_tau: e_0 = log g(h0);  e_{t+1} = v[t] - scum[t].
// Pass A: per-(b,h,chunk) streaming (max M, scaled sum S) summary.
// ---------------------------------------------------------------------------
__global__ void scan_chunks(const float* __restrict__ vout, const float* __restrict__ scum,
                            float* __restrict__ msum, float* __restrict__ ssum)
{
    int id = blockIdx.x * blockDim.x + threadIdx.x;       // Bb*CHUNKS*Hh
    if (id >= Bb * CHUNKS * Hh) return;
    int h = id & (Hh - 1);
    int c = (id >> 9) & (CHUNKS - 1);
    int b = id >> 15;
    float M = -INFINITY, S = 0.f;
    size_t base = ((size_t)b * Tt + (size_t)c * CLEN) * Hh + h;   // coalesced in h
    for (int j = 0; j < CLEN; ++j) {
        float u = vout[base] - scum[base];
        if (u <= M) S += expf(u - M);
        else { S = S * expf(M - u) + 1.f; M = u; }
        base += Hh;
    }
    size_t o = ((size_t)b * Hh + h) * CHUNKS + c;
    msum[o] = M; ssum[o] = S;
}

// Pass B: exclusive combine of chunk summaries per (b,h), seeded with e_0.
__global__ void scan_prefix(const float* __restrict__ h0,
                            float* __restrict__ msum, float* __restrict__ ssum)
{
    int id = blockIdx.x * blockDim.x + threadIdx.x;       // Bb*Hh = 8192
    if (id >= Bb * Hh) return;
    float M = log_gf(h0[id]);   // h0 is [B,1,H] -> flat b*H+h
    float S = 1.f;
    size_t o = (size_t)id * CHUNKS;
    for (int c = 0; c < CHUNKS; ++c) {
        float mc = msum[o + c], sc = ssum[o + c];
        msum[o + c] = M; ssum[o + c] = S;                 // write exclusive prefix
        if (mc <= M) S += sc * expf(mc - M);
        else { S = S * expf(M - mc) + sc; M = mc; }
    }
}

// Pass C: replay each chunk with its prefix; out[t] = exp(scum[t] + LSE_{t+1}).
// Overwrites v in-place in d_out (each thread reads then writes its own range).
__global__ void scan_apply(const float* __restrict__ scum,
                           const float* __restrict__ msum, const float* __restrict__ ssum,
                           float* __restrict__ vout)
{
    int id = blockIdx.x * blockDim.x + threadIdx.x;
    if (id >= Bb * CHUNKS * Hh) return;
    int h = id & (Hh - 1);
    int c = (id >> 9) & (CHUNKS - 1);
    int b = id >> 15;
    size_t o = ((size_t)b * Hh + h) * CHUNKS + c;
    float M = msum[o], S = ssum[o];
    size_t base = ((size_t)b * Tt + (size_t)c * CLEN) * Hh + h;
    for (int j = 0; j < CLEN; ++j) {
        float sc = scum[base];
        float u = vout[base] - sc;
        if (u <= M) S += expf(u - M);
        else { S = S * expf(M - u) + 1.f; M = u; }
        vout[base] = expf(sc + M + logf(S));
        base += Hh;
    }
}

// ---------------------------------------------------------------------------
extern "C" void kernel_launch(void* const* d_in, const int* in_sizes, int n_in,
                              void* d_out, int out_size, void* d_ws, size_t ws_size,
                              hipStream_t stream)
{
    (void)in_sizes; (void)n_in; (void)out_size; (void)ws_size;
    const float* x  = (const float*)d_in[0];
    const float* h0 = (const float*)d_in[1];
    const float* Wf = (const float*)d_in[2];
    const float* bf = (const float*)d_in[3];
    const float* Wi = (const float*)d_in[4];
    const float* bi = (const float*)d_in[5];
    const float* Wh = (const float*)d_in[6];
    const float* bh = (const float*)d_in[7];
    float* out = (float*)d_out;

    // Workspace layout: scum [Mrows*Hh] f32 | Wt hi | Wt lo | msum | ssum
    char* ws = (char*)d_ws;
    float*  scum = (float*)ws;
    __bf16* whi  = (__bf16*)(ws + (size_t)Mrows * Hh * 4);
    __bf16* wlo  = (__bf16*)((char*)whi + (size_t)3 * Dd * Hh * 2);
    float*  msum = (float*)((char*)wlo + (size_t)3 * Dd * Hh * 2);
    float*  ssum = msum + (size_t)Bb * Hh * CHUNKS;

    prep_w<<<(3 * Dd * Hh + 255) / 256, 256, 0, stream>>>(Wf, Wi, Wh, whi, wlo);
    gemm_act<<<Mrows / 16, 256, 67200, stream>>>(x, whi, wlo, bf, bi, bh, out, scum);
    const int nthr = Bb * CHUNKS * Hh;
    scan_chunks<<<nthr / 256, 256, 0, stream>>>(out, scum, msum, ssum);
    scan_prefix<<<(Bb * Hh) / 256, 256, 0, stream>>>(h0, msum, ssum);
    scan_apply<<<nthr / 256, 256, 0, stream>>>(scum, msum, ssum, out);
}
